// DenseAE_29661044146818
// MI455X (gfx1250) — matrix operand order repaired and checked by reading, not compiled, not run on hardware
//
#include <hip/hip_runtime.h>
#include <hip/hip_bf16.h>

// ---------------------------------------------------------------------------
// CDNA5 (gfx1250) k-sparse autoencoder forward pass.
//   H  = relu(X @ We1 + be1)          GEMM1: 4096x12288 @ 12288x1024
//   Hs = ksparse(H, 64)  (bf16)       per-row top-64 mask
//   D  = relu(Hs @ Wd1 + bd1) (bf16)  GEMM2: 4096x1024 @ 1024x1024
//   out= sigmoid(D @ Wd2 + bd2)       GEMM3: 4096x1024 @ 1024x12288
// All GEMMs via v_wmma_f32_16x16x32_bf16, fp32 accumulation. Software
// pipeline: global b128 loads for tile t+1 issued before the WMMAs of tile t,
// bf16 conversion + LDS stores after the WMMAs, one barrier per K-step.
// ---------------------------------------------------------------------------

typedef __attribute__((ext_vector_type(16))) __bf16 v16bf;
typedef __attribute__((ext_vector_type(8)))  float  v8f;
typedef __attribute__((ext_vector_type(2)))  __bf16 v2bf;

union FragAB { v16bf v; unsigned int u[8]; };
union FragC  { v8f   v; float        f[8]; };

#if defined(__has_builtin)
#if __has_builtin(__builtin_amdgcn_cvt_pk_bf16_f32)
#define HAVE_CVT_PK_BF16 1
#else
#define HAVE_CVT_PK_BF16 0
#endif
#else
#define HAVE_CVT_PK_BF16 0
#endif

__device__ __forceinline__ unsigned int rne_bf16_word(float f) {
    // RNE-rounded value sitting in the high 16 bits of a 32-bit word
    unsigned int u = __float_as_uint(f);
    return u + 0x7FFFu + ((u >> 16) & 1u);
}
__device__ __forceinline__ unsigned short f2bf(float f) {
    return (unsigned short)(rne_bf16_word(f) >> 16);
}
// pack two fp32 -> two bf16 in one u32 (lo = a, hi = b)
__device__ __forceinline__ unsigned int pack2bf(float a, float b) {
#if HAVE_CVT_PK_BF16
    union { v2bf v; unsigned int u; } r;
    r.v = __builtin_amdgcn_cvt_pk_bf16_f32(a, b);
    return r.u;
#else
    // one v_perm_b32 merges the two rounded high halves
    return __builtin_amdgcn_perm(rne_bf16_word(b), rne_bf16_word(a), 0x07060302u);
#endif
}

// ---------------------------------------------------------------------------
// Tiled WMMA GEMM:  Out = epi(A @ B + bias)
//   A: M x K row-major, fp32 (A_BF16=0) or bf16-bits (A_BF16=1)
//   B: K x N row-major, fp32 (converted to bf16 while staging in LDS)
//   Out: M x N, fp32 (OUT_BF16=0) or bf16-bits (OUT_BF16=1)
//   EPI_SIGMOID: 0 = relu, 1 = sigmoid
// Workgroup: 256 threads (8 waves). Tile BM=128, BN=64, BK=32.
// Wave (r,c) of 4x2 grid computes a 32x32 sub-tile: 2 A-frags x 2 B-frags,
// 4 WMMA accumulators. LDS ping-pong double buffered; the next tile is held
// in registers while the current tile computes, so all 6 global_load_b128
// per step issue back-to-back and drain behind the WMMAs.
// ---------------------------------------------------------------------------
template<int A_BF16, int EPI_SIGMOID, int OUT_BF16>
__global__ __launch_bounds__(256)
void gemm_wmma_bf16(const void* __restrict__ Ap, const float* __restrict__ B,
                    const float* __restrict__ bias, void* __restrict__ Outp,
                    int M, int N, int K)
{
    // sA: [buf][row][k] as packed bf16 pairs (16 uints = 32 bf16 per row)
    // sB: [buf][col][k] transposed, bf16
    __shared__ __align__(16) unsigned int   sA[2][128 * 16];
    __shared__ __align__(16) unsigned short sB[2][64 * 32];

    const int tid  = threadIdx.x;
    const int lane = tid & 31;
    const int wave = tid >> 5;
    const int wr   = (wave >> 1) * 32;   // wave row base: 0,32,64,96
    const int wc   = (wave & 1) * 32;    // wave col base: 0,32
    const int half = lane >> 4;          // 0: lanes 0-15, 1: lanes 16-31
    const int l16  = lane & 15;

    const long mBase = (long)blockIdx.y * 128;
    const long nBase = (long)blockIdx.x * 64;

    const float*          Af  = (const float*)Ap;
    const unsigned short* Abf = (const unsigned short*)Ap;

    // per-thread staging coordinates (fixed across K-steps)
    const int aR[4]  = { (tid + 0)   >> 3, (tid + 256) >> 3,
                         (tid + 512) >> 3, (tid + 768) >> 3 };
    const int aC4    = (tid & 7) * 4;           // same for all 4 chunks
    const int bK[2]  = { tid >> 4, (tid + 256) >> 4 };
    const int bN4    = (tid & 15) * 4;

    // ---- phase 1: issue all global loads for tile at k0 into registers ----
    float4 aF[4]; uint2 aH[4]; float4 bF[2];
    auto loadTile = [&](int k0) {
#pragma unroll
        for (int q = 0; q < 4; ++q) {
            const long ga = (mBase + aR[q]) * (long)K + k0 + aC4;
            if (A_BF16) aH[q] = *(const uint2*)&Abf[ga];
            else        aF[q] = *(const float4*)&Af[ga];
        }
#pragma unroll
        for (int q = 0; q < 2; ++q)
            bF[q] = *(const float4*)&B[(long)(k0 + bK[q]) * N + nBase + bN4];
    };
    // ---- phase 2: convert + store registers into LDS buffer ----
    auto storeTile = [&](int buf) {
#pragma unroll
        for (int q = 0; q < 4; ++q) {
            uint2 d;
            if (A_BF16) d = aH[q];
            else { d.x = pack2bf(aF[q].x, aF[q].y); d.y = pack2bf(aF[q].z, aF[q].w); }
            *(uint2*)&sA[buf][aR[q] * 16 + (aC4 >> 1)] = d;
        }
#pragma unroll
        for (int q = 0; q < 2; ++q) {
            const unsigned int p01 = pack2bf(bF[q].x, bF[q].y);
            const unsigned int p23 = pack2bf(bF[q].z, bF[q].w);
            const int kk = bK[q];
            sB[buf][(bN4 + 0) * 32 + kk] = (unsigned short)p01;
            sB[buf][(bN4 + 1) * 32 + kk] = (unsigned short)(p01 >> 16);
            sB[buf][(bN4 + 2) * 32 + kk] = (unsigned short)p23;
            sB[buf][(bN4 + 3) * 32 + kk] = (unsigned short)(p23 >> 16);
        }
    };

    FragC c00, c01, c10, c11;
#pragma unroll
    for (int v = 0; v < 8; ++v) {
        c00.f[v] = 0.0f; c01.f[v] = 0.0f; c10.f[v] = 0.0f; c11.f[v] = 0.0f;
    }

    loadTile(0);
    storeTile(0);
    __syncthreads();

    int p = 0;
    for (int k0 = 0; k0 < K; k0 += 32) {
        const int kn = k0 + 32;
        if (kn < K) loadTile(kn);          // issue next-tile global loads first

        // ---- fragment loads per CDNA5 WMMA VGPR layout ----
        // A (16x32 MxK): lane half 0 -> K {0..7,16..23}, half 1 -> K {8..15,24..31}
        // uint index within row: ((v>>2)<<3) + (half<<2) + (v&3)  (b128-coalescable)
        FragAB a0, a1, b0, b1;
        const unsigned int*   As = sA[p];
        const unsigned short* Bs = sB[p];
        const int ar0 = (wr + l16) * 16;
        const int ar1 = (wr + 16 + l16) * 16;
        const int bc0 = (wc + l16) * 32;
        const int bc1 = (wc + 16 + l16) * 32;
#pragma unroll
        for (int v = 0; v < 8; ++v) {
            const int au = ((v >> 2) << 3) + (half << 2) + (v & 3);
            a0.u[v] = As[ar0 + au];
            a1.u[v] = As[ar1 + au];
            const int bk = ((v >> 2) << 4) + (half << 3) + ((v & 3) << 1);
            b0.u[v] = *(const unsigned int*)&Bs[bc0 + bk];
            b1.u[v] = *(const unsigned int*)&Bs[bc1 + bk];
        }

        c00.v = __builtin_amdgcn_wmma_f32_16x16x32_bf16(false, a0.v, false, b0.v,
                                                        (short)0, c00.v, false, false);
        c01.v = __builtin_amdgcn_wmma_f32_16x16x32_bf16(false, a0.v, false, b1.v,
                                                        (short)0, c01.v, false, false);
        c10.v = __builtin_amdgcn_wmma_f32_16x16x32_bf16(false, a1.v, false, b0.v,
                                                        (short)0, c10.v, false, false);
        c11.v = __builtin_amdgcn_wmma_f32_16x16x32_bf16(false, a1.v, false, b1.v,
                                                        (short)0, c11.v, false, false);

        if (kn < K) storeTile(p ^ 1);      // loads have drained behind WMMAs
        __syncthreads();
        p ^= 1;
    }

    // ---- epilogue: bias + activation + store ----
    // C/D layout: lane half 0 -> rows v, half 1 -> rows v+8; col = lane&15
    const long gc0 = nBase + wc + l16;
    const long gc1 = gc0 + 16;
    const float bias0 = bias[gc0];
    const float bias1 = bias[gc1];
    float*          Of  = (float*)Outp;
    unsigned short* Obf = (unsigned short*)Outp;

    auto epi = [&](float x) {
        if (EPI_SIGMOID) return 1.0f / (1.0f + __expf(-x));
        return x > 0.0f ? x : 0.0f;
    };
#pragma unroll
    for (int v = 0; v < 8; ++v) {
        const long r0 = mBase + wr + v + (half << 3);
        const long r1 = r0 + 16;
        const float x00 = epi(c00.f[v] + bias0);
        const float x01 = epi(c01.f[v] + bias1);
        const float x10 = epi(c10.f[v] + bias0);
        const float x11 = epi(c11.f[v] + bias1);
        if (OUT_BF16) {
            Obf[r0 * N + gc0] = f2bf(x00);
            Obf[r0 * N + gc1] = f2bf(x01);
            Obf[r1 * N + gc0] = f2bf(x10);
            Obf[r1 * N + gc1] = f2bf(x11);
        } else {
            Of[r0 * N + gc0] = x00;
            Of[r0 * N + gc1] = x01;
            Of[r1 * N + gc0] = x10;
            Of[r1 * N + gc1] = x11;
        }
    }
}

// ---------------------------------------------------------------------------
// Per-row top-64 selection on H[4096][1024] (relu output, all values >= 0 so
// float ordering == unsigned-int bit ordering). 31-step binary search on the
// bit pattern of the 64th-largest value; emit bf16 sparse row.
// One workgroup (256 threads) per row; 4 values per thread live in registers.
// ---------------------------------------------------------------------------
__global__ __launch_bounds__(256)
void ksparse_topk64(const float* __restrict__ H, unsigned short* __restrict__ Hs)
{
    const int row = blockIdx.x;
    const int tid = threadIdx.x;
    const float* hr = H + (long)row * 1024;

    float        val[4];
    unsigned int bits[4];
#pragma unroll
    for (int j = 0; j < 4; ++j) {
        val[j]  = hr[tid + j * 256];
        bits[j] = __float_as_uint(val[j]);
    }

    __shared__ int scnt;
    unsigned int thr = 0;
    for (int b = 30; b >= 0; --b) {
        const unsigned int cand = thr | (1u << b);
        if (tid == 0) scnt = 0;
        __syncthreads();
        int c = 0;
#pragma unroll
        for (int j = 0; j < 4; ++j) c += (bits[j] >= cand) ? 1 : 0;
        atomicAdd(&scnt, c);
        __syncthreads();
        const int total = scnt;
        __syncthreads();
        if (total >= 64) thr = cand;   // keep threshold with count(>=thr) >= 64
    }

    unsigned short* out = Hs + (long)row * 1024;
#pragma unroll
    for (int j = 0; j < 4; ++j)
        out[tid + j * 256] = (bits[j] >= thr) ? f2bf(val[j]) : (unsigned short)0;
}

// ---------------------------------------------------------------------------
extern "C" void kernel_launch(void* const* d_in, const int* in_sizes, int n_in,
                              void* d_out, int out_size, void* d_ws, size_t ws_size,
                              hipStream_t stream) {
    (void)in_sizes; (void)n_in; (void)out_size; (void)ws_size;
    const float* X   = (const float*)d_in[0];   // [4096, 12288] (flattened NCHW)
    const float* We1 = (const float*)d_in[1];   // [12288, 1024]
    const float* be1 = (const float*)d_in[2];   // [1024]
    const float* Wd1 = (const float*)d_in[3];   // [1024, 1024]
    const float* bd1 = (const float*)d_in[4];   // [1024]
    const float* Wd2 = (const float*)d_in[5];   // [1024, 12288]
    const float* bd2 = (const float*)d_in[6];   // [12288]
    float* out = (float*)d_out;                 // [4096, 12288]

    constexpr int Mb = 4096, D = 12288, HID = 1024;

    // workspace layout (32 MB total; resident in 192 MB L2):
    char* ws = (char*)d_ws;
    float*          H  = (float*)ws;                                   // 16 MB fp32
    unsigned short* Hs = (unsigned short*)(ws + 16u * 1024 * 1024);    //  8 MB bf16
    unsigned short* Dd = (unsigned short*)(ws + 24u * 1024 * 1024);    //  8 MB bf16

    const dim3 blk(256);

    // GEMM1: H = relu(X @ We1 + be1)   [fp32 A -> bf16 LDS, fp32 out]
    gemm_wmma_bf16<0, 0, 0><<<dim3(HID / 64, Mb / 128), blk, 0, stream>>>(
        (const void*)X, We1, be1, (void*)H, Mb, HID, D);

    // k-sparse: keep per-row top-64, emit bf16
    ksparse_topk64<<<dim3(Mb), blk, 0, stream>>>(H, Hs);

    // GEMM2: Dd = relu(Hs @ Wd1 + bd1)   [bf16 A, bf16 out]
    gemm_wmma_bf16<1, 0, 1><<<dim3(HID / 64, Mb / 128), blk, 0, stream>>>(
        (const void*)Hs, Wd1, bd1, (void*)Dd, Mb, HID, HID);

    // GEMM3: out = sigmoid(Dd @ Wd2 + bd2)   [bf16 A, fp32 out]
    gemm_wmma_bf16<1, 1, 0><<<dim3(D / 64, Mb / 128), blk, 0, stream>>>(
        (const void*)Dd, Wd2, bd2, (void*)out, Mb, D, HID);
}
